// AttentionLayer_57732950393486
// MI455X (gfx1250) — matrix operand order repaired and checked by reading, not compiled, not run on hardware
//
#include <hip/hip_runtime.h>

// ---------------------------------------------------------------------------
// Channel attention (B=8, C=256, N=128*128) for MI455X (gfx1250, wave32).
// All GEMMs: v_wmma_f32_16x16x32_bf16 (fp32 accum), 2x2 register tiles per
// wave.  bf16 operands that need no conversion are staged with
// GLOBAL_LOAD_ASYNC_TO_LDS_B128 (ASYNCcnt); fp32 operands are converted to
// bf16 during LDS staging.  K of the 16384-deep reduction is split 8-ways
// into deterministic partial buffers (no atomics).
// ---------------------------------------------------------------------------

typedef __attribute__((ext_vector_type(16))) __bf16 v16bf;
typedef __attribute__((ext_vector_type(8)))  float  v8f;

#define BATCH  8
#define CCH    256
#define NPIX   16384
#define KSPLIT 8
#define KCHUNK (NPIX / KSPLIT)   // 2048

// round-to-nearest-even fp32 -> bf16 (raw u16 in low bits)
__device__ __forceinline__ unsigned int f2bf(float f) {
  unsigned int u = __float_as_uint(f);
  return (u + 0x7FFFu + ((u >> 16) & 1u)) >> 16;
}
__device__ __forceinline__ unsigned int pack2(float a, float b) {
  return f2bf(a) | (f2bf(b) << 16);
}

union FragU { uint4 u[2]; v16bf v; };

// A-matrix 16x32 bf16 fragment from row-major LDS tile (ldk = K stride).
// Lane L: m = L&15, kb = (L<16)?0:8 ; e[0..7]=A[m][kb..], e[8..15]=A[m][16+kb..].
__device__ __forceinline__ v16bf load_a_frag(const unsigned short* lds, int row0,
                                             int ldk, int kofs, int lane) {
  const int m  = lane & 15;
  const int kb = (lane < 16) ? 0 : 8;
  const unsigned short* p = lds + (row0 + m) * ldk + kofs + kb;
  FragU f;
  f.u[0] = *(const uint4*)(p);
  f.u[1] = *(const uint4*)(p + 16);
  return f.v;
}

// B-matrix 32x16 bf16 fragment from K-transposed LDS tile lds[n][k].
// Lane L: n = L&15, kb = (L<16)?0:8 ; e[0..7]=B[kb..][n], e[8..15]=B[16+kb..][n].
__device__ __forceinline__ v16bf load_b_frag(const unsigned short* lds, int col0,
                                             int ldk, int kofs, int lane) {
  const int n  = lane & 15;
  const int kb = (lane < 16) ? 0 : 8;
  const unsigned short* p = lds + (col0 + n) * ldk + kofs + kb;
  FragU f;
  f.u[0] = *(const uint4*)(p);
  f.u[1] = *(const uint4*)(p + 16);
  return f.v;
}

#define WMMA_BF16(A, B, C) \
  __builtin_amdgcn_wmma_f32_16x16x32_bf16(false, (A), false, (B), (short)0, (C), false, false)

// ---------------------------------------------------------------------------
// Stage 1: pk = Wk*x + bk ; pv = Wv*x + bv     (per batch: 256x16384, K=256)
// Block tile 128(M ch) x 64(N pix), BK=32; wave tile 32x32 for BOTH outputs
// (8 wmma per step, B fragments shared between K and V).
// ---------------------------------------------------------------------------
__global__ __launch_bounds__(256) void proj_kernel(
    const float* __restrict__ x, const float* __restrict__ kw,
    const float* __restrict__ kbias, const float* __restrict__ vw,
    const float* __restrict__ vbias,
    unsigned short* __restrict__ pk, unsigned short* __restrict__ pv) {
  __shared__ __align__(16) unsigned short sAk[128 * 32];  // [m][k]
  __shared__ __align__(16) unsigned short sAv[128 * 32];  // [m][k]
  __shared__ __align__(16) unsigned short sBt[64 * 32];   // [n][k]

  const int b    = blockIdx.y;
  const int tile = blockIdx.x;            // (256/128) * (16384/64) = 512
  const int m0   = (tile & 1) * 128;
  const int n0   = (tile >> 1) * 64;
  const int t    = threadIdx.x;
  const int lane = t & 31;
  const int wave = t >> 5;
  const int wm   = wave >> 1;             // 0..3 -> M offset wm*32
  const int wn   = wave & 1;              // 0..1 -> N offset wn*32

  const int ar = t >> 1;                  // 0..127 (weight row), 16 floats each
  const int ak = (t & 1) * 16;
  const int bk = t >> 3;                  // 0..31  (x k-row), 8 pixels each
  const int bn = (t & 7) * 8;

  v8f accK[2][2] = {}; v8f accV[2][2] = {};

  for (int k0 = 0; k0 < CCH; k0 += 32) {
    // weights fp32 -> bf16, row-major
    {
      const float* pK = kw + (m0 + ar) * CCH + k0 + ak;
      const float* pV = vw + (m0 + ar) * CCH + k0 + ak;
      float4 a0 = *(const float4*)(pK);      float4 a1 = *(const float4*)(pK + 4);
      float4 a2 = *(const float4*)(pK + 8);  float4 a3 = *(const float4*)(pK + 12);
      float4 c0 = *(const float4*)(pV);      float4 c1 = *(const float4*)(pV + 4);
      float4 c2 = *(const float4*)(pV + 8);  float4 c3 = *(const float4*)(pV + 12);
      uint4 qa0, qa1, qc0, qc1;
      qa0.x = pack2(a0.x, a0.y); qa0.y = pack2(a0.z, a0.w);
      qa0.z = pack2(a1.x, a1.y); qa0.w = pack2(a1.z, a1.w);
      qa1.x = pack2(a2.x, a2.y); qa1.y = pack2(a2.z, a2.w);
      qa1.z = pack2(a3.x, a3.y); qa1.w = pack2(a3.z, a3.w);
      qc0.x = pack2(c0.x, c0.y); qc0.y = pack2(c0.z, c0.w);
      qc0.z = pack2(c1.x, c1.y); qc0.w = pack2(c1.z, c1.w);
      qc1.x = pack2(c2.x, c2.y); qc1.y = pack2(c2.z, c2.w);
      qc1.z = pack2(c3.x, c3.y); qc1.w = pack2(c3.z, c3.w);
      *(uint4*)&sAk[ar * 32 + ak]     = qa0;
      *(uint4*)&sAk[ar * 32 + ak + 8] = qa1;
      *(uint4*)&sAv[ar * 32 + ak]     = qc0;
      *(uint4*)&sAv[ar * 32 + ak + 8] = qc1;
    }
    // x tile fp32 -> bf16, transposed into [n][k]
    {
      const float* pX = x + ((size_t)b * CCH + (k0 + bk)) * NPIX + n0 + bn;
      float4 v0 = *(const float4*)(pX);
      float4 v1 = *(const float4*)(pX + 4);
      sBt[(bn + 0) * 32 + bk] = (unsigned short)f2bf(v0.x);
      sBt[(bn + 1) * 32 + bk] = (unsigned short)f2bf(v0.y);
      sBt[(bn + 2) * 32 + bk] = (unsigned short)f2bf(v0.z);
      sBt[(bn + 3) * 32 + bk] = (unsigned short)f2bf(v0.w);
      sBt[(bn + 4) * 32 + bk] = (unsigned short)f2bf(v1.x);
      sBt[(bn + 5) * 32 + bk] = (unsigned short)f2bf(v1.y);
      sBt[(bn + 6) * 32 + bk] = (unsigned short)f2bf(v1.z);
      sBt[(bn + 7) * 32 + bk] = (unsigned short)f2bf(v1.w);
    }
    __syncthreads();
    v16bf aK0 = load_a_frag(sAk, wm * 32,      32, 0, lane);
    v16bf aK1 = load_a_frag(sAk, wm * 32 + 16, 32, 0, lane);
    v16bf aV0 = load_a_frag(sAv, wm * 32,      32, 0, lane);
    v16bf aV1 = load_a_frag(sAv, wm * 32 + 16, 32, 0, lane);
    v16bf b0  = load_b_frag(sBt, wn * 32,      32, 0, lane);
    v16bf b1  = load_b_frag(sBt, wn * 32 + 16, 32, 0, lane);
    accK[0][0] = WMMA_BF16(aK0, b0, accK[0][0]);
    accK[0][1] = WMMA_BF16(aK0, b1, accK[0][1]);
    accK[1][0] = WMMA_BF16(aK1, b0, accK[1][0]);
    accK[1][1] = WMMA_BF16(aK1, b1, accK[1][1]);
    accV[0][0] = WMMA_BF16(aV0, b0, accV[0][0]);
    accV[0][1] = WMMA_BF16(aV0, b1, accV[0][1]);
    accV[1][0] = WMMA_BF16(aV1, b0, accV[1][0]);
    accV[1][1] = WMMA_BF16(aV1, b1, accV[1][1]);
    __syncthreads();
  }

  const int rb = (lane < 16) ? 0 : 8;
  const int dl = lane & 15;
#pragma unroll
  for (int mi = 0; mi < 2; ++mi)
#pragma unroll
    for (int ni = 0; ni < 2; ++ni)
#pragma unroll
      for (int r = 0; r < 8; ++r) {
        const int c = m0 + wm * 32 + mi * 16 + rb + r;
        const int n = n0 + wn * 32 + ni * 16 + dl;
        const size_t o = ((size_t)b * CCH + c) * NPIX + n;
        pk[o] = (unsigned short)f2bf(accK[mi][ni][r] + kbias[c]);
        pv[o] = (unsigned short)f2bf(accV[mi][ni][r] + vbias[c]);
      }
}

// ---------------------------------------------------------------------------
// Stage 2: energy[b,c,d] = sum_n pk[b,c,n] * y[b,d,n]  (256x256, K=16384)
// Block tile 128(c) x 64(d), BK=64; wave tile 32x32 (2x2 accumulators).
// K split across gridDim.z into KSPLIT deterministic partial buffers.
// pk (bf16) is copied global->LDS with GLOBAL_LOAD_ASYNC_TO_LDS_B128.
// ---------------------------------------------------------------------------
__global__ __launch_bounds__(256) void energy_kernel(
    const unsigned short* __restrict__ pk, const float* __restrict__ y,
    float* __restrict__ energyP) {
  __shared__ __align__(16) unsigned short sA[128 * 64];   // [c][k]
  __shared__ __align__(16) unsigned short sBt[64 * 64];   // [d][k]

  const int b    = blockIdx.y;
  const int kz   = blockIdx.z;
  const int tile = blockIdx.x;            // 2 * 4 = 8
  const int m0   = (tile & 1) * 128;      // c tile
  const int n0   = (tile >> 1) * 64;      // d tile
  const int t    = threadIdx.x;
  const int lane = t & 31;
  const int wave = t >> 5;
  const int wm   = wave >> 1;             // 0..3 -> c offset wm*32
  const int wn   = wave & 1;              // 0..1 -> d offset wn*32

  const int ar = t >> 1;                  // 0..127 (c row), 32 bf16 each
  const int ak = (t & 1) * 32;
  const int br = t >> 2;                  // 0..63  (d row), 16 floats each
  const int bk = (t & 3) * 16;

  const unsigned short* pkBase =
      pk + ((size_t)b * CCH + (m0 + ar)) * NPIX + (size_t)kz * KCHUNK + ak;
  const float* yBase =
      y + ((size_t)b * CCH + (n0 + br)) * NPIX + (size_t)kz * KCHUNK + bk;
  // LDS byte offset = low 32 bits of the generic pointer (LDS aperture keeps
  // the workgroup-relative offset in addr[31:0]).
  const unsigned ldsA = (unsigned)(uintptr_t)&sA[ar * 64 + ak];

  v8f acc[2][2] = {};

  for (int k0 = 0; k0 < KCHUNK; k0 += 64) {
    // async copy of 64B of bf16 pk row into LDS (ASYNCcnt); INST_OFFSET is
    // applied to both the LDS and global addresses, so one reg pair suffices.
    unsigned long long ga = (unsigned long long)(uintptr_t)(pkBase + k0);
    asm volatile(
        "global_load_async_to_lds_b128 %0, %1, off\n\t"
        "global_load_async_to_lds_b128 %0, %1, off offset:16\n\t"
        "global_load_async_to_lds_b128 %0, %1, off offset:32\n\t"
        "global_load_async_to_lds_b128 %0, %1, off offset:48"
        :: "v"(ldsA), "v"(ga) : "memory");

    // y fp32 -> bf16 into sBt (row-major over k, which is what B-frags want)
    const float4* yv = (const float4*)(yBase + k0);
    float4 y0 = yv[0], y1 = yv[1], y2 = yv[2], y3 = yv[3];
    uint4 q0, q1;
    q0.x = pack2(y0.x, y0.y); q0.y = pack2(y0.z, y0.w);
    q0.z = pack2(y1.x, y1.y); q0.w = pack2(y1.z, y1.w);
    q1.x = pack2(y2.x, y2.y); q1.y = pack2(y2.z, y2.w);
    q1.z = pack2(y3.x, y3.y); q1.w = pack2(y3.z, y3.w);
    *(uint4*)&sBt[br * 64 + bk]     = q0;
    *(uint4*)&sBt[br * 64 + bk + 8] = q1;
    __builtin_prefetch(yBase + k0 + 64, 0, 0);       // global_prefetch_b8

    asm volatile("s_wait_asynccnt 0x0" ::: "memory");
    __syncthreads();

#pragma unroll
    for (int ks = 0; ks < 64; ks += 32) {
      v16bf a0 = load_a_frag(sA,  wm * 32,      64, ks, lane);
      v16bf a1 = load_a_frag(sA,  wm * 32 + 16, 64, ks, lane);
      v16bf b0 = load_b_frag(sBt, wn * 32,      64, ks, lane);
      v16bf b1 = load_b_frag(sBt, wn * 32 + 16, 64, ks, lane);
      acc[0][0] = WMMA_BF16(a0, b0, acc[0][0]);
      acc[0][1] = WMMA_BF16(a0, b1, acc[0][1]);
      acc[1][0] = WMMA_BF16(a1, b0, acc[1][0]);
      acc[1][1] = WMMA_BF16(a1, b1, acc[1][1]);
    }
    __syncthreads();
  }

  float* eout = energyP + (size_t)kz * (BATCH * CCH * CCH) + (size_t)b * CCH * CCH;
  const int rb = (lane < 16) ? 0 : 8;
  const int dl = lane & 15;
#pragma unroll
  for (int mi = 0; mi < 2; ++mi)
#pragma unroll
    for (int ni = 0; ni < 2; ++ni)
#pragma unroll
      for (int r = 0; r < 8; ++r) {
        const int c = m0 + wm * 32 + mi * 16 + rb + r;
        const int d = n0 + wn * 32 + ni * 16 + dl;
        eout[(size_t)c * CCH + d] = acc[mi][ni][r];
      }
}

// ---------------------------------------------------------------------------
// Stage 3: sum the K-chunk partials, row softmax over d, write attention
// TRANSPOSED (attT[d][c]) as bf16. One wave per 256-wide row.
// ---------------------------------------------------------------------------
__global__ __launch_bounds__(256) void softmax_kernel(
    const float* __restrict__ energyP, unsigned short* __restrict__ attT) {
  const int t    = threadIdx.x;
  const int lane = t & 31;
  const int wave = t >> 5;
  const int row  = blockIdx.x * 8 + wave;   // 0..2047 = b*256 + c
  const int b    = row >> 8;
  const int c    = row & 255;

  const float* e = energyP + (size_t)row * CCH;
  float v[8];
#pragma unroll
  for (int j = 0; j < 8; ++j) {
    float s = 0.0f;
#pragma unroll
    for (int z = 0; z < KSPLIT; ++z)
      s += e[(size_t)z * (BATCH * CCH * CCH) + lane + 32 * j];
    v[j] = s;
  }

  float m = v[0];
#pragma unroll
  for (int j = 1; j < 8; ++j) m = fmaxf(m, v[j]);
  for (int off = 16; off > 0; off >>= 1) m = fmaxf(m, __shfl_xor(m, off, 32));

  float s = 0.0f;
#pragma unroll
  for (int j = 0; j < 8; ++j) { v[j] = __expf(v[j] - m); s += v[j]; }
  for (int off = 16; off > 0; off >>= 1) s += __shfl_xor(s, off, 32);
  const float inv = 1.0f / s;

#pragma unroll
  for (int j = 0; j < 8; ++j) {
    const int d = lane + 32 * j;
    attT[((size_t)b * CCH + d) * CCH + c] = (unsigned short)f2bf(v[j] * inv);
  }
}

// ---------------------------------------------------------------------------
// Stage 4: out[b,d,n] = gamma * sum_c attT[b,d,c]*pv[b,c,n] + x[b,d,n]
// Block tile 128(d) x 64(n), BK=32; wave tile 32x32 (2x2 accumulators).
// attT (bf16, pure copy) staged via GLOBAL_LOAD_ASYNC_TO_LDS_B128.
// ---------------------------------------------------------------------------
__global__ __launch_bounds__(256) void out_kernel(
    const unsigned short* __restrict__ attT, const unsigned short* __restrict__ pv,
    const float* __restrict__ x, const float* __restrict__ gamma,
    float* __restrict__ out) {
  __shared__ __align__(16) unsigned short sA[128 * 32];   // [d][c]
  __shared__ __align__(16) unsigned short sBt[64 * 32];   // [n][c]

  const int b    = blockIdx.y;
  const int tile = blockIdx.x;            // (256/128) * (16384/64) = 512
  const int m0   = (tile & 1) * 128;      // d tile
  const int n0   = (tile >> 1) * 64;      // pixel tile
  const int t    = threadIdx.x;
  const int lane = t & 31;
  const int wave = t >> 5;
  const int wm   = wave >> 1;             // 0..3 -> d offset wm*32
  const int wn   = wave & 1;              // 0..1 -> n offset wn*32

  const int ar = t >> 1;                  // 0..127 (d row), 16 bf16 each
  const int ak = (t & 1) * 16;
  const int bc = t >> 3;                  // 0..31 (c row of pv), 8 pixels each
  const int bn = (t & 7) * 8;

  const unsigned short* aBase = attT + ((size_t)b * CCH + m0 + ar) * CCH + ak;
  const unsigned ldsA = (unsigned)(uintptr_t)&sA[ar * 32 + ak];

  v8f acc[2][2] = {};

  for (int k0 = 0; k0 < CCH; k0 += 32) {
    // async copy of 32B of bf16 attT row into LDS
    unsigned long long ga = (unsigned long long)(uintptr_t)(aBase + k0);
    asm volatile(
        "global_load_async_to_lds_b128 %0, %1, off\n\t"
        "global_load_async_to_lds_b128 %0, %1, off offset:16"
        :: "v"(ldsA), "v"(ga) : "memory");

    // pv tile (bf16), transposed into [n][c]
    const unsigned short* pvp =
        pv + ((size_t)b * CCH + k0 + bc) * NPIX + n0 + bn;
    uint4 pw = *(const uint4*)pvp;
    sBt[(bn + 0) * 32 + bc] = (unsigned short)(pw.x & 0xFFFFu);
    sBt[(bn + 1) * 32 + bc] = (unsigned short)(pw.x >> 16);
    sBt[(bn + 2) * 32 + bc] = (unsigned short)(pw.y & 0xFFFFu);
    sBt[(bn + 3) * 32 + bc] = (unsigned short)(pw.y >> 16);
    sBt[(bn + 4) * 32 + bc] = (unsigned short)(pw.z & 0xFFFFu);
    sBt[(bn + 5) * 32 + bc] = (unsigned short)(pw.z >> 16);
    sBt[(bn + 6) * 32 + bc] = (unsigned short)(pw.w & 0xFFFFu);
    sBt[(bn + 7) * 32 + bc] = (unsigned short)(pw.w >> 16);

    asm volatile("s_wait_asynccnt 0x0" ::: "memory");
    __syncthreads();

    v16bf a0 = load_a_frag(sA,  wm * 32,      32, 0, lane);
    v16bf a1 = load_a_frag(sA,  wm * 32 + 16, 32, 0, lane);
    v16bf b0 = load_b_frag(sBt, wn * 32,      32, 0, lane);
    v16bf b1 = load_b_frag(sBt, wn * 32 + 16, 32, 0, lane);
    acc[0][0] = WMMA_BF16(a0, b0, acc[0][0]);
    acc[0][1] = WMMA_BF16(a0, b1, acc[0][1]);
    acc[1][0] = WMMA_BF16(a1, b0, acc[1][0]);
    acc[1][1] = WMMA_BF16(a1, b1, acc[1][1]);
    __syncthreads();
  }

  const float g = gamma[0];
  const int rb = (lane < 16) ? 0 : 8;
  const int dl = lane & 15;
#pragma unroll
  for (int mi = 0; mi < 2; ++mi)
#pragma unroll
    for (int ni = 0; ni < 2; ++ni)
#pragma unroll
      for (int r = 0; r < 8; ++r) {
        const int d = m0 + wm * 32 + mi * 16 + rb + r;
        const int n = n0 + wn * 32 + ni * 16 + dl;
        const size_t o = ((size_t)b * CCH + d) * NPIX + n;
        out[o] = g * acc[mi][ni][r] + x[o];
      }
}

// ---------------------------------------------------------------------------
extern "C" void kernel_launch(void* const* d_in, const int* in_sizes, int n_in,
                              void* d_out, int out_size, void* d_ws, size_t ws_size,
                              hipStream_t stream) {
  (void)in_sizes; (void)n_in; (void)out_size; (void)ws_size;

  const float* x     = (const float*)d_in[0];
  const float* y     = (const float*)d_in[1];
  const float* kw    = (const float*)d_in[2];
  const float* kb    = (const float*)d_in[3];
  const float* vw    = (const float*)d_in[4];
  const float* vb    = (const float*)d_in[5];
  const float* gamma = (const float*)d_in[6];
  float* out = (float*)d_out;

  // workspace: pk (bf16) | pv (bf16) | energy partials (f32 x KSPLIT) | attT
  unsigned short* pk = (unsigned short*)d_ws;
  unsigned short* pv = pk + (size_t)BATCH * CCH * NPIX;
  float* energyP = (float*)(pv + (size_t)BATCH * CCH * NPIX);
  unsigned short* attT =
      (unsigned short*)(energyP + (size_t)KSPLIT * BATCH * CCH * CCH);

  proj_kernel<<<dim3((CCH / 128) * (NPIX / 64), BATCH), 256, 0, stream>>>(
      x, kw, kb, vw, vb, pk, pv);
  energy_kernel<<<dim3((CCH / 128) * (CCH / 64), BATCH, KSPLIT), 256, 0, stream>>>(
      pk, y, energyP);
  softmax_kernel<<<dim3(BATCH * CCH / 8), 256, 0, stream>>>(energyP, attT);
  out_kernel<<<dim3((CCH / 128) * (NPIX / 64), BATCH), 256, 0, stream>>>(
      attT, pv, x, gamma, out);
}
